// PolicyCausalTransformer_58772332479043
// MI455X (gfx1250) — matrix operand order repaired and checked by reading, not compiled, not run on hardware
//
#include <hip/hip_runtime.h>
#include <hip/hip_bf16.h>

typedef __bf16 bf16_t;
typedef __attribute__((ext_vector_type(16))) __bf16 v16bf;
typedef __attribute__((ext_vector_type(8)))  float  v8f;

// ---- problem constants (mirror the reference) ----
constexpr int ONE_STEP  = 288;    // 256 img + 16 text + 8 think + 8 act
constexpr int TOK_TO_AO = 280;    // action-out token offset within a step
constexpr int HIST      = 4;      // attention history length (steps)
constexpr int Q_LEN     = 2304;   // 288 * 8
constexpr int KV_LEN    = 2305;   // Q_LEN + 1 sink
constexpr int NB        = 2;
constexpr int NH        = 16;
constexpr int DH        = 64;

constexpr int BM = 128;           // q rows per workgroup (8 waves x 16)
constexpr int BN = 32;            // kv rows per inner block

// softmax done in exp2 domain: scores pre-scaled by 1/sqrt(64) * log2(e)
#define QSCALE (0.125f * 1.44269504088896340736f)

__global__ __launch_bounds__(256)
void PolicyCausalAttn_gfx1250(const float* __restrict__ Qg,
                              const float* __restrict__ Kg,
                              const float* __restrict__ Vg,
                              float* __restrict__ Og) {
    const int bh   = blockIdx.y;
    const int q0wg = blockIdx.x * BM;
    const int tid  = threadIdx.x;
    const int lane = tid & 31;
    const int wave = tid >> 5;
    const int m    = lane & 15;   // q column (QK) / d row (PV) within tile
    const int hi   = lane >> 4;
    const int hi8  = hi * 8;

    const int qi = q0wg + wave * 16 + m;   // absolute q row owned in column m

    const float* Qb = Qg + (size_t)bh * Q_LEN * DH;
    const float* Kb = Kg + (size_t)bh * KV_LEN * DH;
    const float* Vb = Vg + (size_t)bh * KV_LEN * DH;
    float*       Ob = Og + (size_t)bh * Q_LEN * DH;

    // double-buffered staging: one barrier per iteration
    __shared__ __align__(16) bf16_t ldsK[2][BN * DH];   // [n][d] row major
    __shared__ __align__(16) bf16_t ldsVT[2][DH * BN];  // [d][n] transposed

    // ---- Q tile as B-operand (Q^T), scaled, bf16 ----
    // B layout: lane(col=m, hi) element e  <->  k = 16*hi + e (contiguous d)
    v16bf bq[2];
    {
        const float* qrow = Qb + (size_t)qi * DH;
        #pragma unroll
        for (int h = 0; h < 2; ++h) {
            const float* p = qrow + 32 * h + 16 * hi;
            #pragma unroll
            for (int e = 0; e < 16; ++e)
                bq[h][e] = (bf16_t)(p[e] * QSCALE);
        }
    }

    // ---- q-side mask classification (hoisted per lane) ----
    const int  qs    = qi / ONE_STEP;
    const int  qr    = qi - qs * ONE_STEP;
    const bool q_ao  = qr == TOK_TO_AO;
    const bool q_ra  = qr >  TOK_TO_AO;
    const int  lim   = TOK_TO_AO + (q_ao ? 1 : 0);  // same-step: kr < lim ok

    // ---- workgroup-uniform kv block range (mask window + sink block 0) ----
    const int min_step = q0wg / ONE_STEP;
    const int max_step = (q0wg + BM - 1) / ONE_STEP;
    const int kv_lo_b  = (min_step > HIST) ? (min_step - HIST) * ONE_STEP : 0;
    const int b_start  = ((kv_lo_b + 1) / BN) < 1 ? 1 : ((kv_lo_b + 1) / BN);
    const int b_last   = ((max_step + 1) * ONE_STEP) / BN;

    float Mrun = -__builtin_inff();
    float Lrun = 0.0f;
    v8f acc[4];
    #pragma unroll
    for (int t = 0; t < 4; ++t) acc[t] = {};

    // ---- staging registers (pipelined global->LDS) ----
    const int sn  = tid >> 3;        // 0..31 local kv row this thread stages
    const int sd0 = (tid & 7) * 8;   // 0..56 d-chunk
    float4 kA, kB, vA, vB;

    auto preload = [&](int blk) {
        int row = blk * BN + sn;
        row = row < (KV_LEN - 1) ? row : (KV_LEN - 1);
        const float* kp = Kb + (size_t)row * DH + sd0;
        const float* vp = Vb + (size_t)row * DH + sd0;
        kA = *reinterpret_cast<const float4*>(kp);
        kB = *reinterpret_cast<const float4*>(kp + 4);
        vA = *reinterpret_cast<const float4*>(vp);
        vB = *reinterpret_cast<const float4*>(vp + 4);
    };

    preload(0);

    int buf = 0;
    for (int b = 0;;) {
        const int n0 = b * BN;
        bf16_t* lK = ldsK[buf];
        bf16_t* lV = ldsVT[buf];

        // ---- stage registers -> LDS (K packed b128, V^T scattered b16) ----
        {
            union { uint4 u; bf16_t h[8]; } pk;
            pk.h[0] = (bf16_t)kA.x; pk.h[1] = (bf16_t)kA.y;
            pk.h[2] = (bf16_t)kA.z; pk.h[3] = (bf16_t)kA.w;
            pk.h[4] = (bf16_t)kB.x; pk.h[5] = (bf16_t)kB.y;
            pk.h[6] = (bf16_t)kB.z; pk.h[7] = (bf16_t)kB.w;
            *reinterpret_cast<uint4*>(&lK[sn * DH + sd0]) = pk.u;
            lV[(sd0 + 0) * BN + sn] = (bf16_t)vA.x;
            lV[(sd0 + 1) * BN + sn] = (bf16_t)vA.y;
            lV[(sd0 + 2) * BN + sn] = (bf16_t)vA.z;
            lV[(sd0 + 3) * BN + sn] = (bf16_t)vA.w;
            lV[(sd0 + 4) * BN + sn] = (bf16_t)vB.x;
            lV[(sd0 + 5) * BN + sn] = (bf16_t)vB.y;
            lV[(sd0 + 6) * BN + sn] = (bf16_t)vB.z;
            lV[(sd0 + 7) * BN + sn] = (bf16_t)vB.w;
        }
        __syncthreads();   // LDS block ready (and buf^1 free: see analysis)

        // ---- kick off next block's loads unconditionally (clamped) ----
        const int  bn   = (b == 0) ? b_start : b + 1;
        const bool more = (bn <= b_last);
        preload(more ? bn : b_last);

        // ---- S^T = K * Q^T : two 16x16 tiles (kv rows x q cols) ----
        v8f c[2];
        #pragma unroll
        for (int t = 0; t < 2; ++t) {
            c[t] = {};
            #pragma unroll
            for (int h = 0; h < 2; ++h) {
                // A layout: lane row = 16t+m, elems 0..15 -> d=32h+16hi+0..15
                union { v16bf v; uint4 u[2]; } a;
                const int off = (16 * t + m) * DH + 32 * h + 16 * hi;
                a.u[0] = *reinterpret_cast<const uint4*>(&lK[off]);
                a.u[1] = *reinterpret_cast<const uint4*>(&lK[off + 8]);
                c[t] = __builtin_amdgcn_wmma_f32_16x16x32_bf16(
                           false, a.v, false, bq[h], (short)0, c[t], false, false);
            }
        }

        // ---- mask: incremental kv classification (one scalar div/block) ----
        const int kb0 = n0 - 1;                              // uniform
        const int ks0 = (kb0 >= 0) ? (kb0 / ONE_STEP) : 0;   // SALU
        const int kr0 = kb0 - ks0 * ONE_STEP;                // -1..287
        float sc[2][8];
        #pragma unroll
        for (int t = 0; t < 2; ++t) {
            const int krt = kr0 + 16 * t + hi8;
            #pragma unroll
            for (int r = 0; r < 8; ++r) {
                int kr = krt + r;
                int ks = ks0;
                if (kr >= ONE_STEP) { kr -= ONE_STEP; ks += 1; }
                const int  du      = qs - ks;
                const bool win     = (unsigned)du <= (unsigned)HIST;
                const bool same_ok = (kr < lim) || (q_ra && kr > TOK_TO_AO);
                const bool prev_ok = (kr != TOK_TO_AO);
                const bool f       = (du != 0) ? prev_ok : same_ok;
                const bool ok      = (kr < 0) || (f && win);   // kr<0: sink
                sc[t][r] = ok ? c[t][r] : -1e30f;
            }
        }

        // ---- online softmax, exp2 domain (16 in-lane + xor16 partner) ----
        float tmax = -__builtin_inff();
        #pragma unroll
        for (int t = 0; t < 2; ++t)
            #pragma unroll
            for (int r = 0; r < 8; ++r) tmax = fmaxf(tmax, sc[t][r]);
        tmax = fmaxf(tmax, __shfl_xor(tmax, 16, 32));
        const float Mnew = fmaxf(Mrun, tmax);
        const float fold = __builtin_amdgcn_exp2f(Mrun - Mnew);

        // raw v_exp_f32: underflow flushes to 0, exactly what softmax wants
        float p[2][8];
        #pragma unroll
        for (int t = 0; t < 2; ++t)
            #pragma unroll
            for (int r = 0; r < 8; ++r)
                p[t][r] = __builtin_amdgcn_exp2f(sc[t][r] - Mnew);

        // depth-4 add tree (FP adds are not compiler-reassociable)
        float s8[8];
        #pragma unroll
        for (int r = 0; r < 8; ++r) s8[r] = p[0][r] + p[1][r];
        const float s40 = s8[0] + s8[1], s41 = s8[2] + s8[3];
        const float s42 = s8[4] + s8[5], s43 = s8[6] + s8[7];
        float rs = (s40 + s41) + (s42 + s43);
        rs += __shfl_xor(rs, 16, 32);

        Lrun = fmaf(Lrun, fold, rs);
        Mrun = Mnew;
        #pragma unroll
        for (int t = 0; t < 4; ++t)
            #pragma unroll
            for (int r = 0; r < 8; ++r) acc[t][r] *= fold;

        // ---- assemble P^T as B-operand: one xor16 shuffle + select ----
        // lane hi=0 needs n_rel 0..15 : own t0 (0..7)  + partner t0 (8..15)
        // lane hi=1 needs n_rel 16..31: partner t1 (16..23) + own t1 (24..31)
        v16bf bp;
        #pragma unroll
        for (int e = 0; e < 8; ++e) {
            const float o0 = __shfl_xor(p[0][e], 16, 32);
            const float o1 = __shfl_xor(p[1][e], 16, 32);
            bp[e]     = (bf16_t)(hi ? o1      : p[0][e]);
            bp[e + 8] = (bf16_t)(hi ? p[1][e] : o0);
        }

        // ---- O^T += V^T * P^T : four 16x16 d-tiles, K=32 in one WMMA ----
        #pragma unroll
        for (int t = 0; t < 4; ++t) {
            union { v16bf v; uint4 u[2]; } a;
            const int off = (16 * t + m) * BN + 16 * hi;   // row d, cols n_rel
            a.u[0] = *reinterpret_cast<const uint4*>(&lV[off]);
            a.u[1] = *reinterpret_cast<const uint4*>(&lV[off + 8]);
            acc[t] = __builtin_amdgcn_wmma_f32_16x16x32_bf16(
                         false, a.v, false, bp, (short)0, acc[t], false, false);
        }

        if (!more) break;
        b = bn;
        buf ^= 1;
    }

    // ---- epilogue: O[qi, d] = acc^T / L ; lane(m,hi) owns d = 16t+8hi+0..7
    const float inv = 1.0f / Lrun;
    float* orow = Ob + (size_t)qi * DH;
    #pragma unroll
    for (int t = 0; t < 4; ++t) {
        float4 v0, v1;
        v0.x = acc[t][0] * inv; v0.y = acc[t][1] * inv;
        v0.z = acc[t][2] * inv; v0.w = acc[t][3] * inv;
        v1.x = acc[t][4] * inv; v1.y = acc[t][5] * inv;
        v1.z = acc[t][6] * inv; v1.w = acc[t][7] * inv;
        *reinterpret_cast<float4*>(orow + 16 * t + hi8)     = v0;
        *reinterpret_cast<float4*>(orow + 16 * t + hi8 + 4) = v1;
    }
}

extern "C" void kernel_launch(void* const* d_in, const int* in_sizes, int n_in,
                              void* d_out, int out_size, void* d_ws, size_t ws_size,
                              hipStream_t stream) {
    (void)in_sizes; (void)n_in; (void)out_size; (void)d_ws; (void)ws_size;
    const float* q = (const float*)d_in[0];
    const float* k = (const float*)d_in[1];
    const float* v = (const float*)d_in[2];
    float* out = (float*)d_out;

    dim3 grid(Q_LEN / BM, NB * NH);   // 18 x 32
    dim3 block(256);
    PolicyCausalAttn_gfx1250<<<grid, block, 0, stream>>>(q, k, v, out);
}
